// BinaryApproximateAttention_62947040690319
// MI455X (gfx1250) — 8-bit matrix operand order repaired and checked by reading, compiled, not run on hardware
//
#include <hip/hip_runtime.h>
#include <hip/hip_bf16.h>

typedef __attribute__((ext_vector_type(8))) int v8i;

#define BATCH 2
#define HEADS 12
#define SEQ   1024
#define DIM   64
#define KP    102
#define NROWS (BATCH*HEADS*SEQ)
#define NEL   (NROWS*DIM)

__device__ __forceinline__ int sgn8(float x) {
    return (x > 0.0f) ? 1 : ((x < 0.0f) ? -1 : 0);
}

__global__ __launch_bounds__(256) void pack_signs_kernel(
    const float* __restrict__ q, const float* __restrict__ k,
    unsigned* __restrict__ q_s8, unsigned* __restrict__ k_s8, int n4) {
    int i = blockIdx.x * 256 + threadIdx.x;
    if (i >= n4) return;
    const float4* qv = (const float4*)q;
    const float4* kv = (const float4*)k;
    float4 a = qv[i];
    float4 b = kv[i];
    unsigned pa = ((unsigned)(sgn8(a.x) & 0xFF)) |
                  ((unsigned)(sgn8(a.y) & 0xFF) << 8) |
                  ((unsigned)(sgn8(a.z) & 0xFF) << 16) |
                  ((unsigned)(sgn8(a.w) & 0xFF) << 24);
    unsigned pb = ((unsigned)(sgn8(b.x) & 0xFF)) |
                  ((unsigned)(sgn8(b.y) & 0xFF) << 8) |
                  ((unsigned)(sgn8(b.z) & 0xFF) << 16) |
                  ((unsigned)(sgn8(b.w) & 0xFF) << 24);
    q_s8[i] = pa;
    k_s8[i] = pb;
}

__device__ __forceinline__ v8i load_B(const char* kbase, int kt, int nl, int half) {
    const int4* brow = (const int4*)(kbase + (size_t)(kt * 16 + nl) * DIM);
    int4 b03 = brow[half];
    int4 b47 = brow[2 + half];
    v8i B;
    B[0] = b03.x; B[1] = b03.y; B[2] = b03.z; B[3] = b03.w;
    B[4] = b47.x; B[5] = b47.y; B[6] = b47.z; B[7] = b47.w;
    return B;
}

__global__ __launch_bounds__(256) void BinaryApproximateAttention_62947040690319_kernel(
    const float* __restrict__ q, const float* __restrict__ k,
    const float* __restrict__ v, const float* __restrict__ mask,
    const char* __restrict__ q_s8, const char* __restrict__ k_s8,
    float* __restrict__ out) {

    __shared__ unsigned char score_h[16 * SEQ];
    __shared__ int   hist[16 * 65];
    __shared__ float q_lds[16 * DIM];
    __shared__ unsigned short sel_idx[16 * KP];
    __shared__ float sel_p[16 * KP];
    __shared__ int thr_h[16];
    __shared__ int need_eq[16];

    const int tile = blockIdx.x & 63;
    const int head = blockIdx.x >> 6;
    const int bb   = head / HEADS;
    const size_t R = (size_t)head * SEQ;
    const int q0   = tile * 16;

    const int tid  = threadIdx.x;
    const int lane = tid & 31;
    const int w    = tid >> 5;
    const int half = lane >> 4;
    const int nl   = lane & 15;

    for (int i = tid; i < 16 * 65; i += 256) hist[i] = 0;
    for (int i = tid; i < 16 * DIM; i += 256) q_lds[i] = q[(R + q0) * DIM + i];
    __syncthreads();

    const int4* arow = (const int4*)(q_s8 + (R + q0 + nl) * DIM);
    int4 a03 = arow[half];
    int4 a47 = arow[2 + half];
    v8i A;
    A[0] = a03.x; A[1] = a03.y; A[2] = a03.z; A[3] = a03.w;
    A[4] = a47.x; A[5] = a47.y; A[6] = a47.z; A[7] = a47.w;

    auto consume = [&](const v8i& a0, const v8i& a1, int key0) {
        const int key1 = key0 + 16;
#pragma unroll
        for (int r = 0; r < 8; ++r) {
            const int m = r + 8 * half;
            int h0 = (64 - a0[r]) >> 1;
            int h1 = (64 - a1[r]) >> 1;
            score_h[m * SEQ + key0] = (unsigned char)h0;
            score_h[m * SEQ + key1] = (unsigned char)h1;
            atomicAdd(&hist[m * 65 + h0], 1);
            atomicAdd(&hist[m * 65 + h1], 1);
        }
    };

    const char* kbase  = k_s8 + R * DIM;
    const int   ktbase = w * 8;

    v8i Bc0 = load_B(kbase, ktbase + 0, nl, half);
    v8i Bc1 = load_B(kbase, ktbase + 1, nl, half);
    v8i pv0, pv1;
#pragma unroll
    for (int i = 0; i < 4; ++i) {
        v8i Bn0, Bn1;
        if (i < 3) {
            Bn0 = load_B(kbase, ktbase + 2 * i + 2, nl, half);
            Bn1 = load_B(kbase, ktbase + 2 * i + 3, nl, half);
        }
        v8i z0 = {};
        v8i z1 = {};
        v8i a0 = __builtin_amdgcn_wmma_i32_16x16x64_iu8(
            true, A, true, Bc0, z0, false, false);
        v8i a1 = __builtin_amdgcn_wmma_i32_16x16x64_iu8(
            true, A, true, Bc1, z1, false, false);
        if (i > 0) {
            consume(pv0, pv1, (ktbase + 2 * i - 2) * 16 + nl);
        }
        pv0 = a0;
        pv1 = a1;
        Bc0 = Bn0;
        Bc1 = Bn1;
    }
    consume(pv0, pv1, (ktbase + 6) * 16 + nl);
    __syncthreads();

    if (tid < 16) {
        int cum = 0, hstar = 64, ngt = 0;
        for (int h = 0; h <= 64; ++h) {
            int c = hist[tid * 65 + h];
            if (cum + c >= KP) { hstar = h; ngt = cum; break; }
            cum += c;
        }
        thr_h[tid]   = hstar;
        need_eq[tid] = KP - ngt;
    }
    __syncthreads();

    for (int qq = 0; qq < 2; ++qq) {
        const int qi    = w + qq * 8;
        const int hstar = thr_h[qi];
        const int nEq   = need_eq[qi];
        const unsigned below = (1u << lane) - 1u;

        int nsel = 0, eq_seen = 0;
        for (int c = 0; c < 32; ++c) {
            int key = c * 32 + lane;
            int h   = score_h[qi * SEQ + key];
            bool gt = (h < hstar);
            bool eq = (h == hstar);
            unsigned eqm = __builtin_amdgcn_ballot_w32(eq);
            bool take = gt || (eq && (eq_seen + __popc(eqm & below) < nEq));
            unsigned tm = __builtin_amdgcn_ballot_w32(take);
            if (take) {
                int pos = nsel + __popc(tm & below);
                sel_idx[qi * KP + pos] = (unsigned short)key;
            }
            nsel    += __popc(tm);
            eq_seen += __popc(eqm);
        }

        float s[4];
        float mymax = -3.0e38f;
        const float4* qr = (const float4*)&q_lds[qi * DIM];
#pragma unroll
        for (int t = 0; t < 4; ++t) {
            int j = lane + t * 32;
            if (j < KP) {
                int key = sel_idx[qi * KP + j];
                const float4* kr = (const float4*)(k + (R + key) * DIM);
                float dot = 0.0f;
#pragma unroll
                for (int d4 = 0; d4 < DIM / 4; ++d4) {
                    float4 kv = kr[d4];
                    float4 qv = qr[d4];
                    dot += kv.x * qv.x + kv.y * qv.y + kv.z * qv.z + kv.w * qv.w;
                }
                s[t] = dot * 0.125f + mask[(size_t)bb * SEQ + key];
            } else {
                s[t] = -3.0e38f;
            }
            mymax = fmaxf(mymax, s[t]);
        }
        for (int off = 16; off > 0; off >>= 1)
            mymax = fmaxf(mymax, __shfl_xor(mymax, off, 32));

        float e[4];
        float mysum = 0.0f;
#pragma unroll
        for (int t = 0; t < 4; ++t) {
            int j = lane + t * 32;
            e[t] = (j < KP) ? __expf(s[t] - mymax) : 0.0f;
            mysum += e[t];
        }
        for (int off = 16; off > 0; off >>= 1)
            mysum += __shfl_xor(mysum, off, 32);
        float inv = 1.0f / mysum;
#pragma unroll
        for (int t = 0; t < 4; ++t) {
            int j = lane + t * 32;
            if (j < KP) sel_p[qi * KP + j] = e[t] * inv;
        }
    }
    __syncthreads();

    const int d  = tid & 63;
    const int qg = tid >> 6;
    for (int g = 0; g < 4; ++g) {
        int qi = g * 4 + qg;
        float acc = 0.0f;
        for (int j = 0; j < KP; ++j) {
            float p = sel_p[qi * KP + j];
            int key = sel_idx[qi * KP + j];
            acc += p * v[(R + key) * DIM + d];
        }
        out[(R + q0 + qi) * DIM + d] = acc;
    }
}

extern "C" void kernel_launch(void* const* d_in, const int* in_sizes, int n_in,
                              void* d_out, int out_size, void* d_ws, size_t ws_size,
                              hipStream_t stream) {
    const float* q    = (const float*)d_in[0];
    const float* k    = (const float*)d_in[1];
    const float* v    = (const float*)d_in[2];
    const float* mask = (const float*)d_in[3];
    float* out        = (float*)d_out;

    char* q_s8 = (char*)d_ws;
    char* k_s8 = q_s8 + (size_t)NEL;

    const int n4 = NEL / 4;
    pack_signs_kernel<<<(n4 + 255) / 256, 256, 0, stream>>>(
        q, k, (unsigned*)q_s8, (unsigned*)k_s8, n4);

    BinaryApproximateAttention_62947040690319_kernel<<<NROWS / 16, 256, 0, stream>>>(
        q, k, v, mask, q_s8, k_s8, out);
}
